// SAA_Attention_25228637897579
// MI455X (gfx1250) — matrix operand order repaired and checked by reading, not compiled, not run on hardware
//
#include <hip/hip_runtime.h>
#include <stdint.h>

// Problem constants (match reference)
#define B_DIM 2048
#define H_DIM 4
#define D_DIM 4096
#define M_GRP 8
#define DM    (D_DIM / M_GRP)   // 512

// Clang native ext vectors
typedef __bf16    v16bf __attribute__((ext_vector_type(16)));
typedef float     v8f   __attribute__((ext_vector_type(8)));
typedef uint32_t  u32x4 __attribute__((ext_vector_type(4)));
typedef uint32_t  u32x8 __attribute__((ext_vector_type(8)));
typedef int       i32x4 __attribute__((ext_vector_type(4)));
typedef int       i32x8 __attribute__((ext_vector_type(8)));
typedef float     f32x2 __attribute__((ext_vector_type(2)));
typedef float     f32x4 __attribute__((ext_vector_type(4)));

__device__ __forceinline__ uint32_t pack_bf16x2(float lo, float hi) {
    // round-to-nearest-even f32 -> bf16, packed pair
    uint32_t ulo = __builtin_bit_cast(uint32_t, lo);
    uint32_t uhi = __builtin_bit_cast(uint32_t, hi);
    ulo = (ulo + 0x7FFFu + ((ulo >> 16) & 1u)) >> 16;
    uhi = (uhi + 0x7FFFu + ((uhi >> 16) & 1u)) >> 16;
    return (ulo & 0xFFFFu) | (uhi << 16);
}

// ---------------------------------------------------------------------------
// Kernel 1: fused grouped tiny-linear + group-major scramble -> S (bf16)
// S[bb, dd] with bb = m*256 + b/8, dd = (b%8)*512 + c:
//   S = sum_h x[b, h, m*512 + c] * local_W[m,h] + local_b[m]
// ---------------------------------------------------------------------------
__global__ __launch_bounds__(256) void prep_s(
    const float* __restrict__ x, const float* __restrict__ lw,
    const float* __restrict__ lb, uint32_t* __restrict__ Sb)
{
    const int p  = blockIdx.x * 256 + threadIdx.x;   // pair index, B*D/2 total
    const int bb = p >> 11;                          // / (D/2)
    const int dd = (p & 2047) << 1;
    const int m  = bb >> 8;
    const int q  = bb & 255;
    const int r  = dd >> 9;
    const int c  = dd & 511;
    const int b  = (q << 3) + r;

    const float* xp = x + (size_t)b * (H_DIM * D_DIM) + m * DM + c;
    const float w0 = lw[m * H_DIM + 0], w1 = lw[m * H_DIM + 1];
    const float w2 = lw[m * H_DIM + 2], w3 = lw[m * H_DIM + 3];
    const float bias = lb[m];

    const f32x2 x0 = *(const f32x2*)(xp);
    const f32x2 x1 = *(const f32x2*)(xp + D_DIM);
    const f32x2 x2 = *(const f32x2*)(xp + 2 * D_DIM);
    const f32x2 x3 = *(const f32x2*)(xp + 3 * D_DIM);

    const float v0 = bias + x0.x * w0 + x1.x * w1 + x2.x * w2 + x3.x * w3;
    const float v1 = bias + x0.y * w0 + x1.y * w1 + x2.y * w2 + x3.y * w3;
    Sb[p] = pack_bf16x2(v0, v1);
}

// ---------------------------------------------------------------------------
// Kernel 2: proj_W fp32 -> bf16 (row-major D x D preserved)
// ---------------------------------------------------------------------------
__global__ __launch_bounds__(256) void prep_w(
    const float* __restrict__ W, u32x4* __restrict__ Wb)
{
    const size_t p = (size_t)blockIdx.x * 256 + threadIdx.x;   // 8-element chunk
    const f32x4* src = (const f32x4*)W + p * 2;
    const f32x4 f0 = src[0];
    const f32x4 f1 = src[1];
    u32x4 o;
    o.x = pack_bf16x2(f0.x, f0.y);
    o.y = pack_bf16x2(f0.z, f0.w);
    o.z = pack_bf16x2(f1.x, f1.y);
    o.w = pack_bf16x2(f1.z, f1.w);
    Wb[p] = o;
}

// ---------------------------------------------------------------------------
// TDM: issue one tensor-DMA moving a 128-row x 32-bf16 K-panel into LDS,
// inserting 4 DWORDs of padding every 16 DWORDs (one row) -> LDS stride 40.
// D# fields per CDNA5 ISA ch.8 (group0: count/lds/global/type,
// group1: data_size/padding/tensor dims/tile dims/strides).
// ---------------------------------------------------------------------------
__device__ __forceinline__ void tdm_load_tile(uint64_t gaddr, uint32_t lds_addr,
                                              uint32_t tensor_rows)
{
    u32x4 g0;
    g0.x = 1u;                                                     // count=1 (valid user D#)
    g0.y = lds_addr;                                               // LDS byte address
    g0.z = (uint32_t)gaddr;                                        // global_addr[31:0]
    g0.w = ((uint32_t)(gaddr >> 32) & 0x01FFFFFFu) | (2u << 30);   // ga[56:32] | type=2

    i32x8 g1;
    g1[0] = (int)((1u << 16)      // data_size = 2 bytes
                | (1u << 20)      // pad_enable
                | (3u << 22)      // pad_interval = 16 DWORDs (one 64B row)
                | (3u << 25));    // pad_amount   = 4 DWORDs (8 bf16)
    g1[1] = (int)((uint32_t)D_DIM << 16);        // tensor_dim0[15:0] = 4096
    g1[2] = (int)(tensor_rows << 16);            // tensor_dim1[15:0]
    g1[3] = (int)(32u << 16);                    // tile_dim0 = 32 (K elements)
    g1[4] = 128;                                 // tile_dim1 = 128 rows
    g1[5] = D_DIM;                               // tensor_dim0_stride = 4096
    g1[6] = 0;
    g1[7] = 0;

    const i32x4 z4 = (i32x4){0, 0, 0, 0};        // groups 2/3 unused (2D tile)
#if defined(__clang_major__) && (__clang_major__ >= 23)
    const i32x8 z8 = (i32x8){0, 0, 0, 0, 0, 0, 0, 0};
    __builtin_amdgcn_tensor_load_to_lds(g0, g1, z4, z4, z8, 0);
#else
    __builtin_amdgcn_tensor_load_to_lds(g0, g1, z4, z4, 0);
#endif
}

// ---------------------------------------------------------------------------
// Kernel 3: out = sigmoid(S @ W^T + b) via v_wmma_f32_16x16x32_bf16.
// Block tile 128(M) x 128(N), BK = 32, 8 wave32 per block in a 4x2 wave grid,
// each wave computes 32x64 (2x4 WMMA tiles). Double-buffered LDS filled by
// the Tensor Data Mover (wave 0 issues, TENSORcnt + barrier hand off).
// ---------------------------------------------------------------------------
__global__ __launch_bounds__(256) void gemm_bf16_wmma(
    const uint16_t* __restrict__ Sb, const uint16_t* __restrict__ Wb,
    const float* __restrict__ pb, float* __restrict__ out)
{
    constexpr int LDA   = 40;             // 32 + 8 pad (bf16 elems) per row
    constexpr int KITER = D_DIM / 32;     // 128

    __shared__ __align__(16) uint16_t As[2][128 * LDA];
    __shared__ __align__(16) uint16_t Bs[2][128 * LDA];

    const int tid  = threadIdx.x;
    const int lane = tid & 31;
    const int wave = tid >> 5;       // 0..7
    const int wm   = wave >> 1;      // 0..3  (M group of 32 rows)
    const int wn   = wave & 1;       // 0..1  (N group of 64 cols)
    const int lrow = lane & 15;
    const int lhi  = lane >> 4;      // 0/1

    // TDM source base addresses (advance 64 B per K-step)
    const uint64_t aG = (uint64_t)(uintptr_t)(Sb + (size_t)blockIdx.y * 128 * D_DIM);
    const uint64_t bG = (uint64_t)(uintptr_t)(Wb + (size_t)blockIdx.x * 128 * D_DIM);
    const uint32_t ldsA[2] = { (uint32_t)(uintptr_t)&As[0][0],
                               (uint32_t)(uintptr_t)&As[1][0] };
    const uint32_t ldsB[2] = { (uint32_t)(uintptr_t)&Bs[0][0],
                               (uint32_t)(uintptr_t)&Bs[1][0] };

    v8f acc[2][4];
    #pragma unroll
    for (int i = 0; i < 2; ++i)
        #pragma unroll
        for (int j = 0; j < 4; ++j)
            acc[i][j] = (v8f){0.f, 0.f, 0.f, 0.f, 0.f, 0.f, 0.f, 0.f};

    // prologue: DMA K-panel 0 into buffer 0
    if (wave == 0) {
        tdm_load_tile(aG, ldsA[0], B_DIM);
        tdm_load_tile(bG, ldsB[0], D_DIM);
    }
    __builtin_amdgcn_s_wait_tensorcnt((short)0);
    __syncthreads();

    const int aRow0 = wm * 32;
    const int bRow0 = wn * 64;
    const int kA = lhi * 8;    // A frag: K halves {0..7,16..23} / {8..15,24..31}
    const int kB = lhi * 8;    // B frag: K halves {0..7,16..23} / {8..15,24..31}

    for (int it = 0; it < KITER; ++it) {
        const int  cur  = it & 1;
        const bool more = (it + 1) < KITER;

        // kick off next K-panel DMA into the other buffer while we compute
        if (more && wave == 0) {
            const uint64_t koff = (uint64_t)(it + 1) * 64u;   // 32 bf16 = 64 B
            tdm_load_tile(aG + koff, ldsA[cur ^ 1], B_DIM);
            tdm_load_tile(bG + koff, ldsB[cur ^ 1], D_DIM);
        }

        v16bf af[2], bf[4];
        #pragma unroll
        for (int i = 0; i < 2; ++i) {
            const uint16_t* p = &As[cur][(aRow0 + i * 16 + lrow) * LDA + kA];
            u32x4 lo = *(const u32x4*)p;
            u32x4 hi = *(const u32x4*)(p + 16);
            u32x8 f  = __builtin_shufflevector(lo, hi, 0, 1, 2, 3, 4, 5, 6, 7);
            af[i] = __builtin_bit_cast(v16bf, f);
        }
        #pragma unroll
        for (int j = 0; j < 4; ++j) {
            const uint16_t* p = &Bs[cur][(bRow0 + j * 16 + lrow) * LDA + kB];
            u32x4 lo = *(const u32x4*)p;
            u32x4 hi = *(const u32x4*)(p + 16);
            u32x8 f  = __builtin_shufflevector(lo, hi, 0, 1, 2, 3, 4, 5, 6, 7);
            bf[j] = __builtin_bit_cast(v16bf, f);
        }

        #pragma unroll
        for (int i = 0; i < 2; ++i)
            #pragma unroll
            for (int j = 0; j < 4; ++j)
                acc[i][j] = __builtin_amdgcn_wmma_f32_16x16x32_bf16(
                    false, af[i], false, bf[j], (short)0, acc[i][j], false, false);

        // wave 0: both DMAs for the next panel must have landed; others: no-op
        __builtin_amdgcn_s_wait_tensorcnt((short)0);
        __syncthreads();
    }

    // epilogue: bias + sigmoid, scatter per C/D layout (M = v + 8*lhi, N = lane&15)
    const int rowBase = blockIdx.y * 128 + wm * 32 + lhi * 8;
    const int colBase = blockIdx.x * 128 + wn * 64 + (lane & 15);
    #pragma unroll
    for (int j = 0; j < 4; ++j) {
        const int   col  = colBase + j * 16;
        const float bias = pb[col];
        #pragma unroll
        for (int i = 0; i < 2; ++i) {
            float* op = out + (size_t)(rowBase + i * 16) * D_DIM + col;
            #pragma unroll
            for (int v = 0; v < 8; ++v) {
                const float val = acc[i][j][v] + bias;
                op[(size_t)v * D_DIM] = 1.0f / (1.0f + __expf(-val));
            }
        }
    }
}

// ---------------------------------------------------------------------------
// Host launcher. Workspace layout: [ S bf16 : 16 MiB ][ W bf16 : 32 MiB ]
// ---------------------------------------------------------------------------
extern "C" void kernel_launch(void* const* d_in, const int* in_sizes, int n_in,
                              void* d_out, int out_size, void* d_ws, size_t ws_size,
                              hipStream_t stream) {
    const float* x  = (const float*)d_in[0];
    const float* lw = (const float*)d_in[1];
    const float* lb = (const float*)d_in[2];
    const float* pw = (const float*)d_in[3];
    const float* pb = (const float*)d_in[4];
    float* out = (float*)d_out;

    uint16_t* Sb = (uint16_t*)d_ws;                       // B*D bf16
    uint16_t* Wb = Sb + (size_t)B_DIM * D_DIM;            // D*D bf16

    prep_s<<<(B_DIM * D_DIM / 2) / 256, 256, 0, stream>>>(x, lw, lb, (uint32_t*)Sb);
    prep_w<<<(int)(((size_t)D_DIM * D_DIM / 8) / 256), 256, 0, stream>>>(pw, (u32x4*)Wb);
    gemm_bf16_wmma<<<dim3(D_DIM / 128, B_DIM / 128), 256, 0, stream>>>(Sb, Wb, pb, out);
}
